// Mamba_78924319031925
// MI455X (gfx1250) — compile-verified, matrix operand order checked by reading, not run on hardware
//
#include <hip/hip_runtime.h>
#include <hip/hip_bf16.h>
#include <cstddef>
#include <cstdint>

#define D_MODEL 768
#define N_LAYER 2
#define VOCAB   32000
#define D_STATE 16
#define D_INNER 1536
#define DT_RANK 48
#define D_CONV  4
#define BB      2
#define LL      1024
#define MTOK    (BB * LL)          // 2048 rows

typedef __attribute__((ext_vector_type(16))) __bf16 v16bf;
typedef __attribute__((ext_vector_type(8)))  __bf16 v8bf;
typedef __attribute__((ext_vector_type(8)))  float  v8f;

union FragBF { v16bf v; v8bf h[2]; };

// ---------------------------------------------------------------------------
// WMMA GEMM:  C[M,N] = A[M,K] @ W[N,K]^T (+ addend), bf16 in, fp32 accumulate.
// 256 threads = 8 waves as 2(M) x 4(N); wave tile 64x32; block tile 128x128.
// K staged through LDS in 128x32 tiles, double buffered, filled with
// global_load_async_to_lds_b128 (ASYNCcnt) so copies overlap WMMA math.
// Requires M % 128 == 0, K % 32 == 0. N edge: clamp loads, guard stores.
// ---------------------------------------------------------------------------
#define BK        32
#define ROW_BYTES 64              // 32 bf16 per row
#define ROW_PITCH 80              // padded: 20-bank stride -> conflict-free b128
#define TILE_LDS  (128 * ROW_PITCH)

__device__ __forceinline__ unsigned lds_off(const void* p) {
    // LDS aperture: addr[63:32] is the aperture tag, addr[31:0] IS the offset.
    return (unsigned)(uintptr_t)p;
}

__device__ __forceinline__ void async_tile_load(const __bf16* gbase, int block_row,
                                                int row_max, int K, int k,
                                                unsigned lds_base, int tid) {
#pragma unroll
    for (int part = 0; part < 2; ++part) {
        int c     = tid + part * 256;      // 512 x 16B chunks per 128x32 tile
        int row   = c >> 2;
        int chunk = c & 3;
        int grow  = block_row + row;
        if (grow > row_max) grow = row_max;
        unsigned voff = ((unsigned)grow * (unsigned)K + (unsigned)(k + chunk * 8)) * 2u;
        unsigned lds  = lds_base + (unsigned)(row * ROW_PITCH + chunk * 16);
        asm volatile("global_load_async_to_lds_b128 %0, %1, %2"
                     :: "v"(lds), "v"(voff), "s"(gbase)
                     : "memory");
    }
}

__device__ __forceinline__ v16bf lds_frag(const char* tile, int row, int half) {
    const char* p = tile + row * ROW_PITCH + half * 16;
    FragBF f;
    f.h[0] = *(const v8bf*)(p);        // K = half*8 .. +7
    f.h[1] = *(const v8bf*)(p + 32);   // K = 16 + half*8 .. +7
    return f.v;
}

template <bool HAS_ADD>
__global__ void wmma_gemm_bf16(const __bf16* __restrict__ A,
                               const __bf16* __restrict__ W,
                               const float*  __restrict__ addend,
                               float* __restrict__ C,
                               int M, int N, int K) {
    __shared__ __align__(16) char smA[2][TILE_LDS];
    __shared__ __align__(16) char smB[2][TILE_LDS];

    const int tid   = threadIdx.x;
    const int lane  = tid & 31;
    const int wave  = tid >> 5;
    const int waveM = wave >> 2;            // 0..1  -> 64 rows each
    const int waveN = wave & 3;             // 0..3  -> 32 cols each
    const int blockM = blockIdx.y * 128;
    const int blockN = blockIdx.x * 128;
    const int half = lane >> 4;
    const int l16  = lane & 15;

    const int nsteps = K / BK;

    v8f acc[4][2] = {};

    // prologue: stage 0
    async_tile_load(A, blockM, M - 1, K, 0, lds_off(&smA[0][0]), tid);
    async_tile_load(W, blockN, N - 1, K, 0, lds_off(&smB[0][0]), tid);

    for (int ks = 0; ks < nsteps; ++ks) {
        if (ks + 1 < nsteps) {
            const int nb = (ks + 1) & 1;
            async_tile_load(A, blockM, M - 1, K, (ks + 1) * BK,
                            lds_off(&smA[nb][0]), tid);
            async_tile_load(W, blockN, N - 1, K, (ks + 1) * BK,
                            lds_off(&smB[nb][0]), tid);
            asm volatile("s_wait_asynccnt 0x4" ::: "memory");   // stage ks done
        } else {
            asm volatile("s_wait_asynccnt 0x0" ::: "memory");
        }
        __syncthreads();                    // all waves' stage-ks data visible

        const char* tA = &smA[ks & 1][0];
        const char* tB = &smB[ks & 1][0];

        v16bf afrag[4], bfrag[2];
#pragma unroll
        for (int i = 0; i < 4; ++i)
            afrag[i] = lds_frag(tA, waveM * 64 + i * 16 + l16, half);
#pragma unroll
        for (int j = 0; j < 2; ++j)
            bfrag[j] = lds_frag(tB, waveN * 32 + j * 16 + l16, half);

#pragma unroll
        for (int i = 0; i < 4; ++i)
#pragma unroll
            for (int j = 0; j < 2; ++j)
                acc[i][j] = __builtin_amdgcn_wmma_f32_16x16x32_bf16(
                    false, afrag[i], false, bfrag[j],
                    (short)0, acc[i][j], false, false);

        __syncthreads();                    // done reading before buffer reuse
    }

#pragma unroll
    for (int i = 0; i < 4; ++i) {
#pragma unroll
        for (int j = 0; j < 2; ++j) {
            int n = blockN + waveN * 32 + j * 16 + l16;
            if (n < N) {
                int mbase = blockM + waveM * 64 + i * 16 + half * 8;
#pragma unroll
                for (int r = 0; r < 8; ++r) {
                    size_t idx = (size_t)(mbase + r) * N + n;
                    float v = acc[i][j][r];
                    if (HAS_ADD) v += addend[idx];
                    C[idx] = v;
                }
            }
        }
    }
}

// ---------------------------------------------------------------------------

__global__ void cvt_f32_bf16(const float* __restrict__ in, __bf16* __restrict__ out,
                             long n) {
    long i = (long)blockIdx.x * blockDim.x + threadIdx.x;
    if (i < n) out[i] = (__bf16)in[i];
}

__global__ void embed_kernel(const int* __restrict__ ids,
                             const float* __restrict__ emb,
                             float* __restrict__ x) {
    long idx = (long)blockIdx.x * blockDim.x + threadIdx.x;   // B*L*D_MODEL
    if (idx >= (long)MTOK * D_MODEL) return;
    int i  = (int)(idx % D_MODEL);
    int bl = (int)(idx / D_MODEL);
    x[idx] = emb[(size_t)ids[bl] * D_MODEL + i];
}

// One block per token row; fp32 in, bf16 out (feeds WMMA).
__global__ void rmsnorm_bf16_kernel(const float* __restrict__ x,
                                    const float* __restrict__ w,
                                    __bf16* __restrict__ out, int D) {
    __shared__ float red[8];
    const int row = blockIdx.x;
    const float* xr = x + (size_t)row * D;
    float s = 0.f;
    for (int i = threadIdx.x; i < D; i += blockDim.x) { float v = xr[i]; s += v * v; }
#pragma unroll
    for (int off = 16; off >= 1; off >>= 1) s += __shfl_xor(s, off, 32);
    const int wid = threadIdx.x >> 5, lane = threadIdx.x & 31;
    if (lane == 0) red[wid] = s;
    __syncthreads();
    if (wid == 0) {
        float t = (lane < 8) ? red[lane] : 0.f;
#pragma unroll
        for (int off = 4; off >= 1; off >>= 1) t += __shfl_xor(t, off, 32);
        if (lane == 0) red[0] = t;
    }
    __syncthreads();
    const float scale = rsqrtf(red[0] / (float)D + 1e-5f);
    for (int i = threadIdx.x; i < D; i += blockDim.x)
        out[(size_t)row * D + i] = (__bf16)(xr[i] * scale * w[i]);
}

// Causal depthwise conv (kernel 4) + bias + SiLU; reads xs half of xz.
__global__ void conv_silu_kernel(const float* __restrict__ xz,
                                 const float* __restrict__ cw,
                                 const float* __restrict__ cb,
                                 float* __restrict__ u,
                                 __bf16* __restrict__ ubf) {
    long idx = (long)blockIdx.x * blockDim.x + threadIdx.x;   // B*L*D_INNER
    if (idx >= (long)MTOK * D_INNER) return;
    int d  = (int)(idx % D_INNER);
    int bl = (int)(idx / D_INNER);
    int l = bl % LL, b = bl / LL;
    float acc = cb[d];
#pragma unroll
    for (int j = 0; j < D_CONV; ++j) {
        int ls = l - (D_CONV - 1) + j;
        if (ls >= 0)
            acc += cw[d * D_CONV + j] * xz[((size_t)(b * LL + ls)) * (2 * D_INNER) + d];
    }
    float sv = acc / (1.f + __expf(-acc));
    u[idx]   = sv;
    ubf[idx] = (__bf16)sv;
}

// delta = softplus(x_dbl[:, :48] @ dt_proj_w^T + dt_proj_b)  (K=48: plain VALU)
__global__ void dt_kernel(const float* __restrict__ xdbl,
                          const float* __restrict__ dtw,
                          const float* __restrict__ dtb,
                          float* __restrict__ delta) {
    long idx = (long)blockIdx.x * blockDim.x + threadIdx.x;   // B*L*D_INNER
    if (idx >= (long)MTOK * D_INNER) return;
    int d  = (int)(idx % D_INNER);
    int bl = (int)(idx / D_INNER);
    const float* xr = xdbl + (size_t)bl * (DT_RANK + 2 * D_STATE);
    const float* wr = dtw + (size_t)d * DT_RANK;
    float acc = dtb[d];
#pragma unroll 8
    for (int r = 0; r < DT_RANK; ++r) acc += xr[r] * wr[r];
    delta[idx] = (acc > 20.f) ? acc : log1pf(__expf(acc));
}

// Selective scan: 16 lanes per (b,d) channel (one per state n), sequential in L.
// Fused epilogue: y = (sum_n h*C + u*D) * silu(res), emitted as bf16.
__global__ void scan_kernel(const float* __restrict__ delta,
                            const float* __restrict__ u,
                            const float* __restrict__ A_log,
                            const float* __restrict__ xdbl,
                            const float* __restrict__ Dp,
                            const float* __restrict__ xz,
                            __bf16* __restrict__ ybf) {
    int tid = blockIdx.x * blockDim.x + threadIdx.x;
    int ch  = tid >> 4;                 // over B*D_INNER channels
    int n   = tid & 15;
    if (ch >= BB * D_INNER) return;
    int b = ch / D_INNER, d = ch % D_INNER;
    const float An = -__expf(A_log[d * D_STATE + n]);
    const float Dd = Dp[d];
    float h = 0.f;
    for (int l = 0; l < LL; ++l) {
        size_t bl = (size_t)b * LL + l;
        float dt = delta[bl * D_INNER + d];
        float uu = u[bl * D_INNER + d];
        const float* xd = xdbl + bl * (DT_RANK + 2 * D_STATE);
        float Bn = xd[DT_RANK + n];
        float Cn = xd[DT_RANK + D_STATE + n];
        h = h * __expf(dt * An) + dt * uu * Bn;
        float p = h * Cn;
        p += __shfl_xor(p, 8, 32);
        p += __shfl_xor(p, 4, 32);
        p += __shfl_xor(p, 2, 32);
        p += __shfl_xor(p, 1, 32);
        if (n == 0) {
            float yv = p + uu * Dd;
            float r  = xz[bl * (2 * D_INNER) + D_INNER + d];
            float sr = r / (1.f + __expf(-r));
            ybf[bl * D_INNER + d] = (__bf16)(yv * sr);
        }
    }
}

static inline char* ws_take(char*& p, size_t bytes) {
    char* r = p;
    p += (bytes + 255) & ~(size_t)255;
    return r;
}

extern "C" void kernel_launch(void* const* d_in, const int* in_sizes, int n_in,
                              void* d_out, int out_size, void* d_ws, size_t ws_size,
                              hipStream_t stream) {
    const int*   ids       = (const int*)  d_in[0];
    const float* embedding = (const float*)d_in[1];
    const float* W_in      = (const float*)d_in[2];
    const float* conv_w    = (const float*)d_in[3];
    const float* conv_b    = (const float*)d_in[4];
    const float* x_proj_w  = (const float*)d_in[5];
    const float* dt_proj_w = (const float*)d_in[6];
    const float* dt_proj_b = (const float*)d_in[7];
    const float* A_log     = (const float*)d_in[8];
    const float* D_param   = (const float*)d_in[9];
    const float* W_out     = (const float*)d_in[10];
    const float* norm_w    = (const float*)d_in[11];
    const float* norm_f_w  = (const float*)d_in[12];
    float* logits = (float*)d_out;

    char* p = (char*)d_ws;
    __bf16* embBF   = (__bf16*)ws_take(p, (size_t)VOCAB * D_MODEL * 2);
    __bf16* winBF   = (__bf16*)ws_take(p, (size_t)N_LAYER * 2 * D_INNER * D_MODEL * 2);
    __bf16* xprojBF = (__bf16*)ws_take(p, (size_t)N_LAYER * (DT_RANK + 2 * D_STATE) * D_INNER * 2);
    __bf16* woutBF  = (__bf16*)ws_take(p, (size_t)N_LAYER * D_MODEL * D_INNER * 2);
    float*  x0      = (float*) ws_take(p, (size_t)MTOK * D_MODEL * 4);
    float*  x1      = (float*) ws_take(p, (size_t)MTOK * D_MODEL * 4);
    __bf16* xnBF    = (__bf16*)ws_take(p, (size_t)MTOK * D_MODEL * 2);
    float*  xz      = (float*) ws_take(p, (size_t)MTOK * 2 * D_INNER * 4);
    float*  uF      = (float*) ws_take(p, (size_t)MTOK * D_INNER * 4);
    __bf16* uBF     = (__bf16*)ws_take(p, (size_t)MTOK * D_INNER * 2);
    float*  xdbl    = (float*) ws_take(p, (size_t)MTOK * (DT_RANK + 2 * D_STATE) * 4);
    float*  dlt     = (float*) ws_take(p, (size_t)MTOK * D_INNER * 4);
    __bf16* yBF     = (__bf16*)ws_take(p, (size_t)MTOK * D_INNER * 2);
    (void)ws_size; (void)n_in; (void)in_sizes; (void)out_size;

    const int T = 256;
    auto gridFor = [](long n, int t) { return (unsigned)((n + t - 1) / t); };

    // One-time (per call) bf16 conversions of GEMM operands.
    { long n = (long)VOCAB * D_MODEL;
      cvt_f32_bf16<<<gridFor(n, T), T, 0, stream>>>(embedding, embBF, n); }
    { long n = (long)N_LAYER * 2 * D_INNER * D_MODEL;
      cvt_f32_bf16<<<gridFor(n, T), T, 0, stream>>>(W_in, winBF, n); }
    { long n = (long)N_LAYER * (DT_RANK + 2 * D_STATE) * D_INNER;
      cvt_f32_bf16<<<gridFor(n, T), T, 0, stream>>>(x_proj_w, xprojBF, n); }
    { long n = (long)N_LAYER * D_MODEL * D_INNER;
      cvt_f32_bf16<<<gridFor(n, T), T, 0, stream>>>(W_out, woutBF, n); }

    // Embedding gather -> residual stream x0 (fp32).
    { long n = (long)MTOK * D_MODEL;
      embed_kernel<<<gridFor(n, T), T, 0, stream>>>(ids, embedding, x0); }

    float* xcur  = x0;
    float* xnext = x1;

    for (int i = 0; i < N_LAYER; ++i) {
        const __bf16* winL   = winBF   + (size_t)i * 2 * D_INNER * D_MODEL;
        const __bf16* xprojL = xprojBF + (size_t)i * (DT_RANK + 2 * D_STATE) * D_INNER;
        const __bf16* woutL  = woutBF  + (size_t)i * D_MODEL * D_INNER;

        // RMSNorm -> bf16 activations
        rmsnorm_bf16_kernel<<<MTOK, T, 0, stream>>>(xcur, norm_w + (size_t)i * D_MODEL,
                                                    xnBF, D_MODEL);

        // xz = xn @ W_in^T   [2048 x 768] x [3072 x 768]^T
        { dim3 g((2 * D_INNER) / 128, MTOK / 128);
          wmma_gemm_bf16<false><<<g, T, 0, stream>>>(xnBF, winL, nullptr, xz,
                                                     MTOK, 2 * D_INNER, D_MODEL); }

        // causal depthwise conv + SiLU -> u (fp32 + bf16)
        { long n = (long)MTOK * D_INNER;
          conv_silu_kernel<<<gridFor(n, T), T, 0, stream>>>(
              xz, conv_w + (size_t)i * D_INNER * D_CONV,
              conv_b + (size_t)i * D_INNER, uF, uBF); }

        // x_dbl = u @ x_proj^T   [2048 x 1536] x [80 x 1536]^T
        { dim3 g(1, MTOK / 128);
          wmma_gemm_bf16<false><<<g, T, 0, stream>>>(uBF, xprojL, nullptr, xdbl,
                                                     MTOK, DT_RANK + 2 * D_STATE, D_INNER); }

        // delta = softplus(x_dbl[:, :48] @ dt_proj^T + b)
        { long n = (long)MTOK * D_INNER;
          dt_kernel<<<gridFor(n, T), T, 0, stream>>>(
              xdbl, dt_proj_w + (size_t)i * D_INNER * DT_RANK,
              dt_proj_b + (size_t)i * D_INNER, dlt); }

        // selective scan + gated epilogue -> y (bf16)
        { long n = (long)BB * D_INNER * 16;
          scan_kernel<<<gridFor(n, T), T, 0, stream>>>(
              dlt, uF, A_log + (size_t)i * D_INNER * D_STATE, xdbl,
              D_param + (size_t)i * D_INNER, xz, yBF); }

        // x_next = y @ W_out^T + x (fused residual)
        { dim3 g(D_MODEL / 128, MTOK / 128);
          wmma_gemm_bf16<true><<<g, T, 0, stream>>>(yBF, woutL, xcur, xnext,
                                                    MTOK, D_MODEL, D_INNER); }

        float* t = xcur; xcur = xnext; xnext = t;
    }

    // final RMSNorm -> bf16
    rmsnorm_bf16_kernel<<<MTOK, T, 0, stream>>>(xcur, norm_f_w, xnBF, D_MODEL);

    // logits = xn @ embedding^T   [2048 x 768] x [32000 x 768]^T
    { dim3 g(VOCAB / 128, MTOK / 128);
      wmma_gemm_bf16<false><<<g, T, 0, stream>>>(xnBF, embBF, nullptr, logits,
                                                 MTOK, VOCAB, D_MODEL); }
}
